// HungarianMatcher_81844896793017
// MI455X (gfx1250) — matrix operand order repaired and checked by reading, not compiled, not run on hardware
//
#include <hip/hip_runtime.h>
#include <hip/hip_bf16.h>
#include <math.h>

typedef __attribute__((ext_vector_type(16))) _Float16 v16h;
typedef __attribute__((ext_vector_type(8)))  float    v8f;
typedef unsigned int u32x4 __attribute__((ext_vector_type(4)));
typedef int          i32x8 __attribute__((ext_vector_type(8)));
typedef int          i32x4 __attribute__((ext_vector_type(4)));

#define BS    32
#define NQ    900
#define NC    1203
#define NT    100
#define QTILE 16
#define NQT   57          // ceil(900/16)
#define NTT   7           // 7*16 = 112 >= 100 target columns
#define KSTEP 32
#define NSTEP ((NC + KSTEP - 1) / KSTEP)   // 38

// focal class-cost term: pos - neg (ALPHA=0.25, GAMMA=2)
__device__ __forceinline__ float focal_term(float x) {
  float p   = 1.0f / (1.0f + __expf(-x));
  float pos = 0.25f * (1.0f - p) * (1.0f - p) * (-__logf(p + 1e-8f));
  float neg = 0.75f * p * p * (-__logf(1.0f - p + 1e-8f));
  return pos - neg;
}

// ---------------------------------------------------------------------------
// TDM: async-copy a 16-row x 32-col f32 tile of the logits row-block into LDS,
// with 1 dword of LDS padding inserted after every 32 dwords (row pitch 33) to
// avoid bank conflicts on the A-fragment gather.  D# layout per CDNA5 ISA ch.8.
// This toolchain exposes the 6-arg builtin form (clang-23 / therock-10.0).
// ---------------------------------------------------------------------------
__device__ __forceinline__ void tdm_load_tile(unsigned lds_off,
                                              unsigned long long gaddr,
                                              int remCols, int remRows) {
  u32x4 g0;
  g0[0] = 1u;                                   // count=1, user descriptor
  g0[1] = lds_off;                              // LDS byte address
  g0[2] = (unsigned)gaddr;                      // global_addr[31:0]
  g0[3] = (unsigned)(gaddr >> 32) | 0x80000000u; // global_addr[56:32] | type=2
  i32x8 g1;
  g1[0] = (int)((2u << 16)      // data_size = 4 bytes
              | (1u << 20)      // pad_enable
              | (4u << 22));    // pad_interval code 4 = every 32 dwords (+1 dw)
  g1[1] = (int)(((unsigned)remCols) << 16);     // tensor_dim0[15:0]
  g1[2] = (int)(((unsigned)remRows) << 16);     // tensor_dim1[15:0]
  g1[3] = (int)(32u << 16);                     // tile_dim0 = 32
  g1[4] = 16;                                   // tile_dim1 = 16, tile_dim2 = 0
  g1[5] = NC;                                   // tensor_dim0_stride[31:0]
  g1[6] = 0;
  g1[7] = 0;
  i32x4 gz4; gz4[0] = gz4[1] = gz4[2] = gz4[3] = 0;   // groups 2/3 unused (2D)
  i32x8 gz8;
  gz8[0] = gz8[1] = gz8[2] = gz8[3] = 0;
  gz8[4] = gz8[5] = gz8[6] = gz8[7] = 0;
  __builtin_amdgcn_tensor_load_to_lds(g0, g1, gz4, gz4, gz8, 0);
}

// ---------------------------------------------------------------------------
// Kernel 1: cost matrix.  One wave32 per (batch, 16-query tile).
// cost_class = focal(logits) @ onehot(labels)^T via v_wmma_f32_16x16x32_f16;
// logits tiles streamed by the Tensor Data Mover, double-buffered in LDS.
// ---------------------------------------------------------------------------
__global__ __launch_bounds__(32)
void cost_kernel(const float* __restrict__ logits,   // [BS][NQ][NC]
                 const float* __restrict__ boxes,    // [BS][NQ][4]
                 const int*   __restrict__ labels,   // [BS][NT]
                 const float* __restrict__ tboxes,   // [BS][NT][4]
                 float* __restrict__ C,              // [BS][NQ][NT]
                 float* __restrict__ Ct)             // [BS][NT][NQ] workspace
{
  __shared__ float  sRaw[2][QTILE][KSTEP + 1];   // TDM-padded row pitch 33
  __shared__ float4 sQB[QTILE];
  __shared__ float4 sTB[NT];
  __shared__ int    sLab[NTT * 16];

  const int lane = threadIdx.x;
  const int b    = blockIdx.y;
  const int q0   = blockIdx.x * QTILE;

  for (int i = lane; i < NTT * 16; i += 32)
    sLab[i] = (i < NT) ? labels[b * NT + i] : -1;      // pad labels -> onehot 0
  for (int i = lane; i < NT; i += 32)
    sTB[i] = ((const float4*)tboxes)[b * NT + i];
  if (lane < QTILE) {
    int q = q0 + lane;
    float4 z = make_float4(0.f, 0.f, 0.f, 0.f);
    sQB[lane] = (q < NQ) ? ((const float4*)boxes)[(size_t)b * NQ + q] : z;
  }
  __syncthreads();

  int labReg[NTT];
#pragma unroll
  for (int tt = 0; tt < NTT; ++tt) labReg[tt] = sLab[tt * 16 + (lane & 15)];

  v8f acc[NTT];
#pragma unroll
  for (int tt = 0; tt < NTT; ++tt)
#pragma unroll
    for (int r = 0; r < 8; ++r) acc[tt][r] = 0.0f;

  const int row  = lane & 15;                 // A fragment: M = lane % 16
  const int kofs = (lane >= 16) ? 16 : 0;     // A: upper half-wave K=16..31
  const int bko  = (lane >= 16) ? 16 : 0;     // B: upper half-wave K=16..31

  const unsigned long long gbase =
      (unsigned long long)(uintptr_t)logits +
      (unsigned long long)((size_t)(b * NQ + q0) * NC) * 4ull;
  const unsigned lds0 = (unsigned)(uintptr_t)(&sRaw[0][0][0]);
  const unsigned lds1 = (unsigned)(uintptr_t)(&sRaw[1][0][0]);
  const int remRows = NQ - q0;

  tdm_load_tile(lds0, gbase, NC, remRows);    // prime the pipeline

  for (int step = 0; step < NSTEP; ++step) {
    const int c0  = step * KSTEP;
    const int cur = step & 1;

    if (step + 1 < NSTEP) {
      // previous use of the other buffer must be fully read before TDM rewrite
      asm volatile("s_wait_dscnt 0x0" ::: "memory");
      tdm_load_tile(cur ? lds0 : lds1,
                    gbase + (unsigned long long)(c0 + KSTEP) * 4ull,
                    NC - (c0 + KSTEP), remRows);
      __builtin_amdgcn_s_wait_tensorcnt(1);   // current tile landed
    } else {
      __builtin_amdgcn_s_wait_tensorcnt(0);
    }
    asm volatile("" ::: "memory");            // LDS reads below see TDM data

    // A fragment (16x32 f16, k = 16*(lane/16) + j, same order as B), focal on the fly
    v16h A = {};
#pragma unroll
    for (int j = 0; j < 16; ++j) {
      int kl = j + kofs;
      A[j] = (_Float16)focal_term(sRaw[cur][row][kl]);
    }

    // one-hot B per 16-target tile, then accumulate
#pragma unroll
    for (int tt = 0; tt < NTT; ++tt) {
      v16h Bm = {};
      const int d = labReg[tt] - c0 - bko;    // compare vs small immediates
#pragma unroll
      for (int j = 0; j < 16; ++j)
        Bm[j] = (d == j) ? (_Float16)1.0f : (_Float16)0.0f;
      acc[tt] = __builtin_amdgcn_wmma_f32_16x16x32_f16(
          false, A, false, Bm, (short)0, acc[tt], false, false);
    }
  }

  // epilogue: fuse bbox L1 + GIoU, write C and transposed Ct
  const int nn   = lane & 15;
  const int mofs = (lane >= 16) ? 8 : 0;
#pragma unroll
  for (int tt = 0; tt < NTT; ++tt) {
#pragma unroll
    for (int r = 0; r < 8; ++r) {
      const int qi = r + mofs;
      const int q  = q0 + qi;
      const int t  = tt * 16 + nn;
      if (q < NQ && t < NT) {
        const float cls = acc[tt][r];
        const float4 qb = sQB[qi];
        const float4 tb = sTB[t];
        const float cb = fabsf(qb.x - tb.x) + fabsf(qb.y - tb.y) +
                         fabsf(qb.z - tb.z) + fabsf(qb.w - tb.w);
        const float ax1 = qb.x - 0.5f * qb.z, ay1 = qb.y - 0.5f * qb.w;
        const float ax2 = qb.x + 0.5f * qb.z, ay2 = qb.y + 0.5f * qb.w;
        const float bx1 = tb.x - 0.5f * tb.z, by1 = tb.y - 0.5f * tb.w;
        const float bx2 = tb.x + 0.5f * tb.z, by2 = tb.y + 0.5f * tb.w;
        const float areaA = (ax2 - ax1) * (ay2 - ay1);
        const float areaB = (bx2 - bx1) * (by2 - by1);
        float iw = fminf(ax2, bx2) - fmaxf(ax1, bx1); iw = fmaxf(iw, 0.f);
        float ih = fminf(ay2, by2) - fmaxf(ay1, by1); ih = fmaxf(ih, 0.f);
        const float inter = iw * ih;
        const float uni   = areaA + areaB - inter;
        const float iou   = inter / uni;
        float hw = fmaxf(ax2, bx2) - fminf(ax1, bx1); hw = fmaxf(hw, 0.f);
        float hh = fmaxf(ay2, by2) - fminf(ay1, by1); hh = fmaxf(hh, 0.f);
        const float hull = hw * hh;
        const float giou = iou - (hull - uni) / hull;
        const float cost = 5.0f * cb + 2.0f * cls - 2.0f * giou;
        C [(size_t)(b * NQ + q) * NT + t] = cost;
        Ct[(size_t)(b * NT + t) * NQ + q] = cost;
      }
    }
  }
}

// ---------------------------------------------------------------------------
// Kernel 2: Jonker-Volgenant LSA on Ct[b] (100 rows x 900 cols), one block per
// batch, 1024 threads (32 wave32s). Columns are thread-owned; argmin reduction
// uses wave shuffles with first-index tie-break to match np.argmin.
// ---------------------------------------------------------------------------
#define LSA_THREADS 1024

__global__ __launch_bounds__(LSA_THREADS)
void lsa_kernel(const float* __restrict__ Ct, float* __restrict__ outIdx)
{
  __shared__ float u[NT + 1];
  __shared__ float v[NQ + 1];
  __shared__ float minv[NQ + 1];
  __shared__ int   p[NQ + 1];
  __shared__ int   way[NQ + 1];
  __shared__ unsigned char used[NQ + 1];
  __shared__ float redV[32];
  __shared__ int   redI[32];
  __shared__ int   s_j0, s_i0;
  __shared__ float s_delta;
  __shared__ int   col4row[NT];

  const int tid = threadIdx.x;
  const int b   = blockIdx.x;
  const float* cost = Ct + (size_t)b * NT * NQ;

  for (int j = tid; j <= NQ; j += LSA_THREADS) { v[j] = 0.f; p[j] = 0; way[j] = 0; }
  if (tid <= NT) u[tid] = 0.f;
  __syncthreads();

  for (int i = 1; i <= NT; ++i) {
    if (tid == 0) { p[0] = i; s_j0 = 0; }
    if (tid <= NQ) { minv[tid] = INFINITY; used[tid] = 0; }
    __syncthreads();

    while (true) {
      if (tid == 0) { used[s_j0] = 1; s_i0 = p[s_j0]; }
      __syncthreads();
      const int i0 = s_i0;
      const int j0 = s_j0;

      if (tid >= 1 && tid <= NQ && !used[tid]) {
        float cur = cost[(size_t)(i0 - 1) * NQ + (tid - 1)] - u[i0] - v[tid];
        if (cur < minv[tid]) { minv[tid] = cur; way[tid] = j0; }
      }
      __syncthreads();

      // argmin over free columns (first-index tie-break)
      float val = (tid >= 1 && tid <= NQ && !used[tid]) ? minv[tid] : INFINITY;
      int   idx = tid;
#pragma unroll
      for (int off = 16; off > 0; off >>= 1) {
        float ov = __shfl_down(val, off, 32);
        int   oi = __shfl_down(idx, off, 32);
        if (ov < val || (ov == val && oi < idx)) { val = ov; idx = oi; }
      }
      const int lane = tid & 31, wv = tid >> 5;
      if (lane == 0) { redV[wv] = val; redI[wv] = idx; }
      __syncthreads();
      if (wv == 0) {
        val = redV[lane]; idx = redI[lane];
#pragma unroll
        for (int off = 16; off > 0; off >>= 1) {
          float ov = __shfl_down(val, off, 32);
          int   oi = __shfl_down(idx, off, 32);
          if (ov < val || (ov == val && oi < idx)) { val = ov; idx = oi; }
        }
        if (lane == 0) { s_delta = val; s_j0 = idx; }
      }
      __syncthreads();

      const float delta = s_delta;
      if (tid <= NQ) {
        if (used[tid]) { u[p[tid]] += delta; v[tid] -= delta; }  // p[] distinct
        else           { minv[tid] -= delta; }
      }
      __syncthreads();
      if (p[s_j0] == 0) break;
    }

    if (tid == 0) {               // augmenting path
      int j0 = s_j0;
      while (j0 != 0) { int j1 = way[j0]; p[j0] = p[j1]; j0 = j1; }
    }
    __syncthreads();
  }

  if (tid >= 1 && tid <= NQ) {
    int pi = p[tid];
    if (pi > 0) col4row[pi - 1] = tid - 1;
  }
  __syncthreads();

  if (tid < NT) {                 // rank == argsort position (values distinct)
    const int myc = col4row[tid];
    int rank = 0;
    for (int t2 = 0; t2 < NT; ++t2) rank += (col4row[t2] < myc) ? 1 : 0;
    outIdx[(size_t)b * NT + rank]                   = (float)myc;  // pred_idx
    outIdx[(size_t)BS * NT + (size_t)b * NT + rank] = (float)tid;  // tgt_idx
  }
}

// ---------------------------------------------------------------------------
extern "C" void kernel_launch(void* const* d_in, const int* in_sizes, int n_in,
                              void* d_out, int out_size, void* d_ws, size_t ws_size,
                              hipStream_t stream) {
  const float* logits = (const float*)d_in[0];   // pred_logits  [32,900,1203]
  const float* boxes  = (const float*)d_in[1];   // pred_boxes   [32,900,4]
  const int*   labels = (const int*)d_in[2];     // tgt_labels   [32,100]
  const float* tboxes = (const float*)d_in[3];   // tgt_boxes    [32,100,4]

  float* C      = (float*)d_out;                       // [32,900,100]
  float* outIdx = C + (size_t)BS * NQ * NT;            // pred_idx | tgt_idx
  float* Ct     = (float*)d_ws;                        // [32,100,900] transpose

  dim3 g1(NQT, BS);
  cost_kernel<<<g1, 32, 0, stream>>>(logits, boxes, labels, tboxes, C, Ct);
  lsa_kernel<<<BS, LSA_THREADS, 0, stream>>>(Ct, outIdx);

  (void)in_sizes; (void)n_in; (void)out_size; (void)ws_size;
}
